// VoxelAttnVFE_44092134261306
// MI455X (gfx1250) — matrix operand order repaired and checked by reading, not compiled, not run on hardware
//
#include <hip/hip_runtime.h>

// ---------------------------------------------------------------------------
// VoxelAttnVFE for MI455X (gfx1250, wave32, WMMA)
//   p    = points @ W_lin                  (K=10: VALU f32)
//   h    = relu(p @ W1 + b1)               (WMMA bf16, f32 acc, fused)
//   attn = sigmoid(h @ W2 + b2)            (WMMA bf16, f32 acc, fused)
//   out  = segment_sum(p * attn, inv)      (f32 atomics)
// ---------------------------------------------------------------------------

#define NPTS 300000
#define INCH 10
#define OUTC 128
#define HIDC 512
#define NVOX 60000

typedef __attribute__((ext_vector_type(16))) __bf16 v16bf;
typedef __attribute__((ext_vector_type(8)))  __bf16 v8bf;
typedef __attribute__((ext_vector_type(8)))  float  v8f;

__device__ __forceinline__ unsigned short f2bfbits(float f) {
  unsigned u = __builtin_bit_cast(unsigned, f);
  u += 0x7FFFu + ((u >> 16) & 1u);           // round-to-nearest-even
  return (unsigned short)(u >> 16);
}
__device__ __forceinline__ float bf2f(unsigned short b) {
  unsigned u = ((unsigned)b) << 16;
  return __builtin_bit_cast(float, u);
}

// wave-private LDS producer->consumer ordering (no workgroup barrier needed:
// each wave uses its own scratch; DS pipe is in-order per wave, we just fence
// the compiler and drain DScnt).
__device__ __forceinline__ void wave_lds_sync() {
  __builtin_amdgcn_wave_barrier();
  asm volatile("s_wait_dscnt 0x0" ::: "memory");
  __builtin_amdgcn_wave_barrier();
}

// Build a 16x32 bf16 A/B fragment from two 8-half (16B) LDS rows.
__device__ __forceinline__ v16bf ldfrag(const unsigned short* lo,
                                        const unsigned short* hi) {
  v8bf a = *(const v8bf*)lo;
  v8bf b = *(const v8bf*)hi;
  v16bf r;
#pragma unroll
  for (int i = 0; i < 8; ++i) { r[i] = a[i]; r[i + 8] = b[i]; }
  return r;
}

// ---------------------------------------------------------------------------
// Prep: repack W1[128,512] and W2[512,128] (f32 row-major) into bf16
// B-fragment order in ws. Fragment (K=32 x N=16): lane holds col N=lane&15,
// K = 16*(lane>>4) + j for halves j=0..15 -> 32 contiguous bytes per lane.
//   W1F[nt(32)][ks(4)][lane(32)][j(16)]   at ws + 0        (65536 halves)
//   W2F[nt(8)][ks(16)][lane(32)][j(16)]   at ws + 65536    (65536 halves)
// ---------------------------------------------------------------------------
__global__ __launch_bounds__(256) void vfe_prep(const float* __restrict__ W1,
                                                const float* __restrict__ W2,
                                                unsigned short* __restrict__ wf) {
  int i = blockIdx.x * 256 + threadIdx.x;          // 0 .. 131071
  if (i < 65536) {
    int j = i & 15, lane = (i >> 4) & 31, ks = (i >> 9) & 3, nt = i >> 11;
    int K = ks * 32 + (lane >> 4) * 16 + j;
    int N = nt * 16 + (lane & 15);
    wf[i] = f2bfbits(W1[K * HIDC + N]);
  } else {
    int loc = i - 65536;
    int j = loc & 15, lane = (loc >> 4) & 31, ks = (loc >> 9) & 15, nt = loc >> 13;
    int K = ks * 32 + (lane >> 4) * 16 + j;
    int N = nt * 16 + (lane & 15);
    wf[65536 + loc] = f2bfbits(W2[K * OUTC + N]);
  }
}

// ---------------------------------------------------------------------------
// Main kernel: 8 waves/block, each wave owns a 16-point M-tile.
// ---------------------------------------------------------------------------
__global__ __launch_bounds__(256) void vfe_main(const float* __restrict__ points,
                                                const float* __restrict__ Wlin,
                                                const float* __restrict__ b1,
                                                const float* __restrict__ b2,
                                                const int*   __restrict__ inv,
                                                const unsigned short* __restrict__ wf,
                                                float* __restrict__ out) {
  __shared__ float sWlin[INCH * OUTC];                                   // 5 KB
  __shared__ __attribute__((aligned(16))) float          sPts[8][16 * INCH];
  __shared__ __attribute__((aligned(16))) unsigned short sP[8][16 * OUTC]; // 32 KB
  __shared__ __attribute__((aligned(16))) unsigned short sH[8][16 * 32];   // 8 KB
  __shared__ int sIdx[8][16];

  const int tid  = threadIdx.x;
  const int w    = tid >> 5;           // wave in block
  const int lane = tid & 31;
  const int nl   = lane & 15;
  const int g    = lane >> 4;

  // cooperative stage of W_lin (f32) once per block
  for (int i = tid; i < INCH * OUTC; i += 256) sWlin[i] = Wlin[i];
  __syncthreads();                     // only workgroup barrier in the kernel

  const int base  = (blockIdx.x * 8 + w) * 16;
  if (base >= NPTS) return;
  const int valid = (NPTS - base < 16) ? (NPTS - base) : 16;

  // stage this wave's 16x10 points tile + voxel ids
  for (int i = lane; i < 16 * INCH; i += 32) {
    int row = i / INCH, col = i % INCH;
    sPts[w][i] = (base + row < NPTS) ? points[(size_t)(base + row) * INCH + col] : 0.f;
  }
  if (lane < 16) sIdx[w][lane] = (base + lane < NPTS) ? inv[base + lane] : 0;
  wave_lds_sync();

  // ---- layer 0: p[16,128] = points @ W_lin (K=10, VALU), stage bf16 to LDS
#pragma unroll
  for (int r = 0; r < 8; ++r) {
    const int M = r + 8 * g;
    float pt[INCH];
#pragma unroll
    for (int k = 0; k < INCH; ++k) pt[k] = sPts[w][M * INCH + k];
#pragma unroll
    for (int t = 0; t < 8; ++t) {
      float acc = 0.f;
#pragma unroll
      for (int k = 0; k < INCH; ++k) acc += pt[k] * sWlin[k * OUTC + t * 16 + nl];
      sP[w][M * OUTC + t * 16 + nl] = f2bfbits(acc);
    }
  }
  wave_lds_sync();

  // p as four 16x32 bf16 A-fragments (kept in VGPRs for all 16 chunks)
  v16bf pA[4];
#pragma unroll
  for (int f = 0; f < 4; ++f)
    pA[f] = ldfrag(&sP[w][nl * OUTC + f * 32 + 16 * g],
                   &sP[w][nl * OUTC + f * 32 + 16 * g + 8]);

  // attn-logit accumulators: 8 N-tiles of 16 -> full 128 outputs
  v8f acc[8];
#pragma unroll
  for (int t = 0; t < 8; ++t)
#pragma unroll
    for (int i = 0; i < 8; ++i) acc[t][i] = 0.f;

  const unsigned short* W1F = wf;
  const unsigned short* W2F = wf + 65536;

  // ---- fused layer1+layer2 over 16 chunks of 32 hidden units each
#pragma unroll 1
  for (int c = 0; c < 16; ++c) {
    const float b1a = b1[c * 32 + nl];
    const float b1b = b1[c * 32 + 16 + nl];
    v8f h0, h1;
#pragma unroll
    for (int i = 0; i < 8; ++i) { h0[i] = b1a; h1[i] = b1b; }

#pragma unroll
    for (int ks = 0; ks < 4; ++ks) {
      v16bf B0 = *(const v16bf*)(W1F + (((2 * c + 0) * 4 + ks) * 32 + lane) * 16);
      v16bf B1 = *(const v16bf*)(W1F + (((2 * c + 1) * 4 + ks) * 32 + lane) * 16);
      h0 = __builtin_amdgcn_wmma_f32_16x16x32_bf16(false, pA[ks], false, B0,
                                                   (short)0, h0, false, false);
      h1 = __builtin_amdgcn_wmma_f32_16x16x32_bf16(false, pA[ks], false, B1,
                                                   (short)0, h1, false, false);
    }

    // relu + re-fragment h[16,32] through wave-private LDS
#pragma unroll
    for (int r = 0; r < 8; ++r) {
      const int M = r + 8 * g;
      sH[w][M * 32 + nl]      = f2bfbits(fmaxf(h0[r], 0.f));
      sH[w][M * 32 + 16 + nl] = f2bfbits(fmaxf(h1[r], 0.f));
    }
    wave_lds_sync();
    v16bf hA = ldfrag(&sH[w][nl * 32 + 16 * g],
                      &sH[w][nl * 32 + 16 * g + 8]);
    wave_lds_sync();   // all lanes done reading before next chunk overwrites

#pragma unroll
    for (int t = 0; t < 8; ++t) {
      v16bf B2 = *(const v16bf*)(W2F + ((t * 16 + c) * 32 + lane) * 16);
      acc[t] = __builtin_amdgcn_wmma_f32_16x16x32_bf16(false, hA, false, B2,
                                                       (short)0, acc[t], false, false);
    }
  }

  // ---- epilogue: sigmoid, gate with p, atomic segment-sum scatter
#pragma unroll
  for (int t = 0; t < 8; ++t) {
    const int n   = t * 16 + nl;
    const float bv = b2[n];
#pragma unroll
    for (int r = 0; r < 8; ++r) {
      const int M = r + 8 * g;
      if (M < valid) {
        float x = acc[t][r] + bv;
        float s = 1.f / (1.f + __expf(-x));
        float pv = bf2f(sP[w][M * OUTC + n]);
        atomicAdd(&out[(size_t)sIdx[w][M] * OUTC + n], pv * s);
      }
    }
  }
}

// ---------------------------------------------------------------------------
extern "C" void kernel_launch(void* const* d_in, const int* in_sizes, int n_in,
                              void* d_out, int out_size, void* d_ws, size_t ws_size,
                              hipStream_t stream) {
  const float* points = (const float*)d_in[0];   // [300000,10]
  const float* W_lin  = (const float*)d_in[1];   // [10,128]
  const float* W1     = (const float*)d_in[2];   // [128,512]
  const float* b1     = (const float*)d_in[3];   // [512]
  const float* W2     = (const float*)d_in[4];   // [512,128]
  const float* b2     = (const float*)d_in[5];   // [128]
  const int*   inv    = (const int*)d_in[6];     // [300000]
  float* out = (float*)d_out;                    // [60000,128]
  unsigned short* wf = (unsigned short*)d_ws;    // 256 KB bf16 weight fragments

  hipMemsetAsync(out, 0, (size_t)out_size * sizeof(float), stream);
  vfe_prep<<<512, 256, 0, stream>>>(W1, W2, wf);
  const int blocks = (NPTS + 128 - 1) / 128;     // 8 waves x 16 points
  vfe_main<<<blocks, 256, 0, stream>>>(points, W_lin, b1, b2, inv, wf, out);
}
